// ChessGNN_77146202570992
// MI455X (gfx1250) — compile-verified, matrix operand order checked by reading, not run on hardware
//
#include <hip/hip_runtime.h>
#include <hip/hip_bf16.h>

typedef _Float16 h16 __attribute__((ext_vector_type(16)));
typedef float    f8  __attribute__((ext_vector_type(8)));

#define NODES 90
#define NPAD  96
#define HD    64
#define HEADS 4
#define FOUT  128
#define NHD   (NODES * HD)          // 5760 floats per board per activation

// ---------------- WMMA fragment helpers (CDNA5 16x16x32 f16 -> f32) ----------------
// A (16x32 f16): lane L holds row m=L&15; lanes 0-15 carry K 0-7 & 16-23,
// lanes 16-31 carry K 8-15 & 24-31 (ISA 7.12.2). Contiguous K -> 2x ds_load_b128.
__device__ __forceinline__ h16 frag_a_ld(const _Float16* p, int ld, int row0, int k0, int lane) {
  int m  = row0 + (lane & 15);
  int kb = k0 + ((lane & 16) ? 8 : 0);
  h16 a;
#pragma unroll
  for (int i = 0; i < 8; ++i) {
    a[i]     = p[m * ld + kb + i];
    a[i + 8] = p[m * ld + kb + 16 + i];
  }
  return a;
}
// B fragment loaded from an N-major ("transposed") buffer pT[n][k] has the exact
// same access pattern as an A fragment with row<->col swapped: contiguous K per lane.
__device__ __forceinline__ h16 frag_bt_ld(const _Float16* pT, int ld, int k0, int col0, int lane) {
  return frag_a_ld(pT, ld, col0, k0, lane);
}
__device__ __forceinline__ f8 wmma16(h16 a, h16 b, f8 c) {
  return __builtin_amdgcn_wmma_f32_16x16x32_f16(false, a, false, b, (short)0, c, false, false);
}

// ---------------- k_init: adjacency bias + zero BN accumulators ----------------
__global__ __launch_bounds__(256) void k_init(float* __restrict__ adjb, float* __restrict__ gacc) {
  int tid = threadIdx.x;
  gacc[tid] = 0.f;                               // 4 x 64 accumulators (sum0,sq0,sum1,sq1)
  for (int idx = tid; idx < NPAD * NPAD; idx += 256) {
    int i = idx / NPAD, j = idx - (idx / NPAD) * NPAD;
    float v = -1e30f;
    if (i < NODES && j < NODES) {
      int yi = i / 9, xi = i - yi * 9, yj = j / 9, xj = j - yj * 9;
      int dx = xi > xj ? xi - xj : xj - xi;
      int dy = yi > yj ? yi - yj : yj - yi;
      if ((yi == yj && dx <= 4) || (xi == xj && dy <= 5)) v = 0.f;
    }
    adjb[idx] = v;
  }
}

// ---------------- k_embed: board planes + pos -> x0 = x @ in_W + in_b ----------------
__global__ __launch_bounds__(256) void k_embed(const float* __restrict__ planes,
                                               const float* __restrict__ in_W,
                                               const float* __restrict__ in_b,
                                               float* __restrict__ x0) {
  __shared__ _Float16 a_s[NPAD * 32];            // 96x32 (row-major), M/K zero-padded
  __shared__ _Float16 bT[HD * 32];               // in_W^T: [n][k], K zero-padded
  __shared__ float    outsT[HD * NPAD];          // result^T: [n][m]
  int b = blockIdx.x, tid = threadIdx.x, lane = tid & 31, wave = tid >> 5;

  // gather planes coalesced over node index
  for (int idx = tid; idx < 32 * NPAD; idx += 256) {
    int k = idx / NPAD, r = idx - k * NPAD;
    float v = 0.f;
    if (r < NODES && k < 16) {
      if (k < 14)       v = planes[(size_t)b * 14 * NODES + k * NODES + r];
      else if (k == 14) v = (float)(r % 9) * (1.f / 9.f);   // gx
      else              v = (float)(r / 9) * (1.f / 10.f);  // gy
    }
    a_s[r * 32 + k] = (_Float16)v;
  }
  for (int idx = tid; idx < 16 * HD; idx += 256) {          // coalesced read of in_W
    int k = idx >> 6, n = idx & 63;
    bT[n * 32 + k] = (_Float16)in_W[idx];
  }
  for (int idx = tid; idx < HD * 16; idx += 256) {          // zero K pad 16..31
    int n = idx >> 4, k = 16 + (idx & 15);
    bT[n * 32 + k] = (_Float16)0.f;
  }
  __syncthreads();

  for (int t = wave; t < 24; t += 8) {           // 6 M-tiles x 4 N-tiles
    int mt = t >> 2, nt = t & 3;
    f8 c = {};
    h16 a  = frag_a_ld(a_s, 32, mt * 16, 0, lane);
    h16 bf = frag_bt_ld(bT, 32, 0, nt * 16, lane);
    c = wmma16(a, bf, c);
    int n  = nt * 16 + (lane & 15);
    int mb = mt * 16 + ((lane & 16) ? 8 : 0);
    float bias = in_b[n];
#pragma unroll
    for (int r = 0; r < 8; ++r) outsT[n * NPAD + mb + r] = c[r] + bias;  // contiguous
  }
  __syncthreads();

  // linear, guard-free, vectorized writeback of rows < 90 (transpose from outsT)
  float4* dst = (float4*)(x0 + (size_t)b * NHD);
  for (int idx = tid; idx < NHD / 4; idx += 256) {
    int r = idx >> 4, c0 = (idx & 15) << 2;
    float4 v;
    v.x = outsT[(c0 + 0) * NPAD + r];
    v.y = outsT[(c0 + 1) * NPAD + r];
    v.z = outsT[(c0 + 2) * NPAD + r];
    v.w = outsT[(c0 + 3) * NPAD + r];
    dst[idx] = v;
  }
}

// ---------------- k_gat: one board per block; h=xW, softmax attn, P@h, BN partials ----
// dynamic LDS layout (bytes):
#define XS_OFF  0            // x  96x64 f16 = 12288   (overlaid by P)
#define WT_OFF  12288        // W^T 64x64 f16 = 8192   (overlaid by P)
#define P_OFF   0            // 4x96x96 f16 = 73728
#define HT_OFF  73728        // h^T 64x96 f16 = 12288
#define SS_OFF  86016        // 4x96 f32 = 1536
#define SD_OFF  87552        // 4x96 f32 = 1536
#define OUT_OFF 89088        // out^T 64x96 f32 = 24576
#define SMEM_GAT 113664

__global__ __launch_bounds__(256) void k_gat(const float* __restrict__ x,
                                             const float* __restrict__ W,
                                             const float* __restrict__ asrc,
                                             const float* __restrict__ adst,
                                             const float* __restrict__ adjb,
                                             float* __restrict__ outraw,
                                             float* __restrict__ gsum,
                                             float* __restrict__ gsumsq) {
  extern __shared__ unsigned char smem[];
  _Float16* xs   = (_Float16*)(smem + XS_OFF);   // [node][chan]  (A operand)
  _Float16* wT   = (_Float16*)(smem + WT_OFF);   // [nchan][kchan] (B operand, N-major)
  _Float16* P    = (_Float16*)(smem + P_OFF);    // [head][i][j]  (A operand)
  _Float16* hT   = (_Float16*)(smem + HT_OFF);   // [chan][node]  (B operand, N-major)
  float* ssrc    = (float*)(smem + SS_OFF);
  float* sdst    = (float*)(smem + SD_OFF);
  float* outsT   = (float*)(smem + OUT_OFF);     // [chan][node]

  int b = blockIdx.x, tid = threadIdx.x, lane = tid & 31, wave = tid >> 5;

  // stage x (f16, rows 90..95 zero) with vectorized global loads
  const float4* xsrc = (const float4*)(x + (size_t)b * NHD);
  for (int idx = tid; idx < (NPAD * HD) / 4; idx += 256) {
    int r = idx >> 4, c0 = (idx & 15) << 2;
    float4 v = {0.f, 0.f, 0.f, 0.f};
    if (r < NODES) v = xsrc[idx];
    _Float16* d = xs + r * HD + c0;
    d[0] = (_Float16)v.x; d[1] = (_Float16)v.y; d[2] = (_Float16)v.z; d[3] = (_Float16)v.w;
  }
  for (int idx = tid; idx < HD * HD; idx += 256) {   // W coalesced read, transposed LDS write
    int k = idx >> 6, n = idx & 63;
    wT[n * HD + k] = (_Float16)W[idx];
  }
  __syncthreads();

  // h = x @ W  (96x64, K=64) -> hT [chan][node]
  for (int t = wave; t < 24; t += 8) {
    int mt = t >> 2, nt = t & 3;
    f8 c = {};
#pragma unroll
    for (int kk = 0; kk < HD; kk += 32) {
      h16 a  = frag_a_ld(xs, HD, mt * 16, kk, lane);
      h16 bf = frag_bt_ld(wT, HD, kk, nt * 16, lane);
      c = wmma16(a, bf, c);
    }
    int n  = nt * 16 + (lane & 15);
    int mb = mt * 16 + ((lane & 16) ? 8 : 0);
    _Float16* d = hT + n * NPAD + mb;              // 8 contiguous halfs per lane
#pragma unroll
    for (int r = 0; r < 8; ++r) d[r] = (_Float16)c[r];
  }
  __syncthreads();

  // per-head attention logits s_src, s_dst
  for (int idx = tid; idx < HEADS * NPAD; idx += 256) {
    int h = idx / NPAD, n = idx - h * NPAD;
    float s1 = 0.f, s2 = 0.f;
#pragma unroll
    for (int d = 0; d < 16; ++d) {
      float v = (float)hT[(h * 16 + d) * NPAD + n];
      s1 += v * asrc[h * 16 + d];
      s2 += v * adst[h * 16 + d];
    }
    ssrc[idx] = s1; sdst[idx] = s2;
  }
  __syncthreads();                 // xs/wT dead from here; P overlays them

  // masked softmax rows -> P[h][i][:] (f16 probabilities, padded with zeros)
  for (int idx = tid; idx < HEADS * NPAD; idx += 256) {
    int h = idx / NPAD, i = idx - h * NPAD;
    _Float16* Pr = P + (size_t)(h * NPAD + i) * NPAD;
    if (i < NODES) {
      float si = ssrc[h * NPAD + i];
      const float* ab = adjb + i * NPAD;
      float mx = -3.0e38f;
      for (int j = 0; j < NODES; ++j) {
        float t = si + sdst[h * NPAD + j];
        t = (t > 0.f) ? t : 0.2f * t;
        t += ab[j];
        mx = fmaxf(mx, t);
      }
      float sum = 0.f;
      for (int j = 0; j < NODES; ++j) {
        float t = si + sdst[h * NPAD + j];
        t = (t > 0.f) ? t : 0.2f * t;
        t += ab[j];
        float e = __expf(t - mx);
        sum += e;
        Pr[j] = (_Float16)e;
      }
      float inv = 1.f / sum;
      for (int j = 0; j < NODES; ++j) Pr[j] = (_Float16)((float)Pr[j] * inv);
      for (int j = NODES; j < NPAD; ++j) Pr[j] = (_Float16)0.f;
    } else {
      for (int j = 0; j < NPAD; ++j) Pr[j] = (_Float16)0.f;
    }
  }
  __syncthreads();

  // out = P @ h per head  (96x64, K=96) -> outsT [chan][node]
  for (int t = wave; t < 24; t += 8) {
    int mt = t >> 2, h = t & 3;
    const _Float16* Ph = P + (size_t)h * NPAD * NPAD;
    f8 c = {};
#pragma unroll
    for (int kt = 0; kt < 3; ++kt) {
      h16 a  = frag_a_ld(Ph, NPAD, mt * 16, kt * 32, lane);
      h16 bf = frag_bt_ld(hT, NPAD, kt * 32, h * 16, lane);
      c = wmma16(a, bf, c);
    }
    int n  = h * 16 + (lane & 15);
    int mb = mt * 16 + ((lane & 16) ? 8 : 0);
    float* d = outsT + n * NPAD + mb;              // 8 contiguous floats per lane
#pragma unroll
    for (int r = 0; r < 8; ++r) d[r] = c[r];
  }
  __syncthreads();

  // coalesced global writeback (transpose) + BN partial sums
  float4* dst = (float4*)(outraw + (size_t)b * NHD);
  for (int idx = tid; idx < NHD / 4; idx += 256) {
    int r = idx >> 4, c0 = (idx & 15) << 2;
    float4 v;
    v.x = outsT[(c0 + 0) * NPAD + r];
    v.y = outsT[(c0 + 1) * NPAD + r];
    v.z = outsT[(c0 + 2) * NPAD + r];
    v.w = outsT[(c0 + 3) * NPAD + r];
    dst[idx] = v;
  }
  if (tid < HD) {
    const float* p = outsT + tid * NPAD;           // contiguous per channel
    float s = 0.f, s2 = 0.f;
    for (int r = 0; r < NODES; ++r) {
      float v = p[r];
      s += v; s2 += v * v;
    }
    atomicAdd(&gsum[tid], s);
    atomicAdd(&gsumsq[tid], s2);
  }
}

// ---------------- k_bn: normalize (training-mode batch stats) + ReLU, in place -------
__global__ __launch_bounds__(256) void k_bn(float4* __restrict__ y,
                                            const float* __restrict__ gsum,
                                            const float* __restrict__ gsumsq,
                                            const float* __restrict__ gamma,
                                            const float* __restrict__ beta,
                                            int total4, float invcnt) {
  int idx = blockIdx.x * 256 + threadIdx.x;
  if (idx < total4) {
    int c0 = (idx << 2) & 63;
    float4 v = y[idx];
    float* pv = &v.x;
#pragma unroll
    for (int j = 0; j < 4; ++j) {
      int c = c0 + j;
      float mean = gsum[c] * invcnt;
      float var  = gsumsq[c] * invcnt - mean * mean;
      float t = (pv[j] - mean) * rsqrtf(var + 1e-5f) * gamma[c] + beta[c];
      pv[j] = fmaxf(t, 0.f);
    }
    y[idx] = v;
  }
}

// ---------------- k_final: node-mean then (16x64)@(64x128) + bias --------------------
__global__ __launch_bounds__(256) void k_final(const float* __restrict__ x,
                                               const float* __restrict__ out_W,
                                               const float* __restrict__ out_b,
                                               float* __restrict__ out) {
  __shared__ _Float16 mean16[16 * HD];           // [batch][chan] (A operand)
  __shared__ _Float16 woutT[FOUT * HD];          // out_W^T [n][k] (B operand, N-major)
  int b0 = blockIdx.x * 16, tid = threadIdx.x, lane = tid & 31, wave = tid >> 5;

  for (int idx = tid; idx < HD * FOUT; idx += 256) {   // coalesced read, transposed write
    int k = idx >> 7, n = idx & 127;
    woutT[n * HD + k] = (_Float16)out_W[idx];
  }
  // node means: one (batch, 4-channel group) per thread, float4 loads
  {
    int bb = tid >> 4, c0 = (tid & 15) << 2;           // 256 threads = 16 x 16 groups
    const float4* p = (const float4*)(x + (size_t)(b0 + bb) * NHD + c0);
    float4 s = {0.f, 0.f, 0.f, 0.f};
    for (int r = 0; r < NODES; ++r) {
      float4 v = p[r * (HD / 4)];
      s.x += v.x; s.y += v.y; s.z += v.z; s.w += v.w;
    }
    _Float16* d = mean16 + bb * HD + c0;
    d[0] = (_Float16)(s.x * (1.f / NODES));
    d[1] = (_Float16)(s.y * (1.f / NODES));
    d[2] = (_Float16)(s.z * (1.f / NODES));
    d[3] = (_Float16)(s.w * (1.f / NODES));
  }
  __syncthreads();

  int nt = wave;                                  // 8 N-tiles of 16
  f8 c = {};
#pragma unroll
  for (int kk = 0; kk < HD; kk += 32) {
    h16 a  = frag_a_ld(mean16, HD, 0, kk, lane);
    h16 bf = frag_bt_ld(woutT, HD, kk, nt * 16, lane);
    c = wmma16(a, bf, c);
  }
  int n  = nt * 16 + (lane & 15);
  int mb = (lane & 16) ? 8 : 0;
  float bias = out_b[n];
#pragma unroll
  for (int r = 0; r < 8; ++r) out[(size_t)(b0 + mb + r) * FOUT + n] = c[r] + bias;
}

// ---------------- host ----------------
extern "C" void kernel_launch(void* const* d_in, const int* in_sizes, int n_in,
                              void* d_out, int out_size, void* d_ws, size_t ws_size,
                              hipStream_t stream) {
  const float* planes = (const float*)d_in[0];
  const float* in_W = (const float*)d_in[1];
  const float* in_b = (const float*)d_in[2];
  const float* W0  = (const float*)d_in[3];
  const float* as0 = (const float*)d_in[4];
  const float* ad0 = (const float*)d_in[5];
  const float* g0  = (const float*)d_in[6];
  const float* b0  = (const float*)d_in[7];
  const float* W1  = (const float*)d_in[8];
  const float* as1 = (const float*)d_in[9];
  const float* ad1 = (const float*)d_in[10];
  const float* g1  = (const float*)d_in[11];
  const float* b1  = (const float*)d_in[12];
  const float* oW  = (const float*)d_in[13];
  const float* ob  = (const float*)d_in[14];

  int B = in_sizes[0] / (14 * NODES);            // 2048

  char* w = (char*)d_ws;
  float* adjb = (float*)w;                       // 96*96*4 = 36864 B
  float* gacc = (float*)(w + 36864);             // 256 f32 BN accumulators
  float* bufX = (float*)(w + 38912);
  size_t per = (size_t)B * NHD;
  float* bufY = bufX + per;

  int total4 = (int)(per / 4);
  int bnb = (total4 + 255) / 256;
  float invcnt = 1.0f / (float)(B * NODES);

  k_init <<<1, 256, 0, stream>>>(adjb, gacc);
  k_embed<<<B, 256, 0, stream>>>(planes, in_W, in_b, bufX);
  k_gat  <<<B, 256, SMEM_GAT, stream>>>(bufX, W0, as0, ad0, adjb, bufY, gacc, gacc + 64);
  k_bn   <<<bnb, 256, 0, stream>>>((float4*)bufY, gacc, gacc + 64, g0, b0, total4, invcnt);
  k_gat  <<<B, 256, SMEM_GAT, stream>>>(bufY, W1, as1, ad1, adjb, bufX, gacc + 128, gacc + 192);
  k_bn   <<<bnb, 256, 0, stream>>>((float4*)bufX, gacc + 128, gacc + 192, g1, b1, total4, invcnt);
  k_final<<<B / 16, 256, 0, stream>>>(bufX, oW, ob, (float*)d_out);
}
